// SeaNet_6442450944220
// MI455X (gfx1250) — matrix operand order repaired and checked by reading, not compiled, not run on hardware
//
#include <hip/hip_runtime.h>
#include <hip/hip_bf16.h>
#include <math.h>

typedef __attribute__((ext_vector_type(16))) _Float16 v16h;
typedef __attribute__((ext_vector_type(8)))  float    v8f;

#define DIVUP(a,b) (((a)+(b)-1)/(b))

// =====================================================================
// Generic wave32 WMMA GEMM:  D[b,m,n] = ep( sum_k A[b,m,k]*B[b,k,n] )
// A elem addr: A + b*sAb + m*sAm + k*sAk
// B elem addr: B + b*sBb + k*sBk + n*sBn
// D elem addr: D + b*sDb + m*sDm + n
// Epilogue (per row m): +bias0, *scale+shift, PReLU(alpha). Any may be null.
// One wave = one 16x16 output tile via v_wmma_f32_16x16x32_f16.
// 4 waves per block (threadIdx.y = consecutive N tiles).
// Out-of-range rows/cols use CLAMPED base pointers and are simply never
// stored (their accumulators hold duplicates of row M-1 / col N-1) — no
// per-element validity selects. The K loop main body has NO masking at
// all (unconditional loads, hoisted per-element offsets); only a single
// remainder step (K%32 != 0) applies k<K zero-selects.
// =====================================================================
__global__ __launch_bounds__(128)
void k_gemm_wmma(const float* __restrict__ A, const float* __restrict__ B,
                 float* __restrict__ D, int M, int N, int K,
                 long sAb, long sAm, long sAk,
                 long sBb, long sBk, long sBn,
                 long sDb, long sDm,
                 const float* __restrict__ bias0,
                 const float* __restrict__ scale,
                 const float* __restrict__ shift,
                 const float* __restrict__ alpha)
{
  const int b    = blockIdx.z;
  const int n0   = (blockIdx.x*4 + threadIdx.y) * 16;
  const int m0   = blockIdx.y * 16;
  if (n0 >= N) return;                 // whole wave exits together
  const int lane = threadIdx.x;        // 0..31
  const int half = lane >> 4;          // lane group
  const int l    = lane & 15;

  const int am  = m0 + l;  const int amc = (am < M) ? am : (M-1);
  const int bn  = n0 + l;  const int bnc = (bn < N) ? bn : (N-1);

  const float* Ak = A + (long)b*sAb + (long)amc*sAm;   // row m (clamped)
  const float* Bk = B + (long)b*sBb + (long)bnc*sBn;   // col n (clamped)

  // loop-invariant per-element K offsets (element -> k index within step)
  // A 16x32: lane group 0 -> K 0..7 & 16..23, group 1 -> K 8..15 & 24..31
  // B 32x16: lane group 0 -> K 0..7 & 16..23, group 1 -> K 8..15 & 24..31
  int  ac[16], bc[16];
  long aoff[16], boff[16];
#pragma unroll
  for (int e = 0; e < 16; ++e) {
    ac[e] = half*8 + e + ((e >= 8) ? 8 : 0);
    bc[e] = half*8 + e + ((e >= 8) ? 8 : 0);
    aoff[e] = (long)ac[e]*sAk;
    boff[e] = (long)bc[e]*sBk;
  }
  const long aStep = 32*sAk, bStep = 32*sBk;

  v8f acc = {0.f,0.f,0.f,0.f,0.f,0.f,0.f,0.f};

  const int Kmain = K & ~31;
  for (int k0 = 0; k0 < Kmain; k0 += 32) {
    v16h af, bf;
#pragma unroll
    for (int e = 0; e < 16; ++e) af[e] = (_Float16)Ak[aoff[e]];
#pragma unroll
    for (int e = 0; e < 16; ++e) bf[e] = (_Float16)Bk[boff[e]];
    acc = __builtin_amdgcn_wmma_f32_16x16x32_f16(false, af, false, bf,
                                                 (short)0, acc, false, false);
    Ak += aStep; Bk += bStep;
  }
  if (Kmain < K) {                      // remainder: zero-pad k >= K
    const int kr = K - Kmain;           // 1..31, wave-uniform
    v16h af, bf;
#pragma unroll
    for (int e = 0; e < 16; ++e) {
      float xv = Ak[(ac[e] < kr) ? aoff[e] : 0];
      af[e] = (_Float16)((ac[e] < kr) ? xv : 0.f);
    }
#pragma unroll
    for (int e = 0; e < 16; ++e) {
      float xv = Bk[(bc[e] < kr) ? boff[e] : 0];
      bf[e] = (_Float16)((bc[e] < kr) ? xv : 0.f);
    }
    acc = __builtin_amdgcn_wmma_f32_16x16x32_f16(false, af, false, bf,
                                                 (short)0, acc, false, false);
  }

  // C/D 16x16 f32: lane holds column n; VGPR r -> row r (+8 for group 1)
  if (bn < N) {
    float* Db = D + (long)b * sDb;
#pragma unroll
    for (int r = 0; r < 8; ++r) {
      int m = m0 + half*8 + r;
      if (m < M) {
        float xv = acc[r];
        if (bias0) xv += bias0[m];
        if (scale) xv = xv * scale[m] + shift[m];
        if (alpha) xv = (xv >= 0.f) ? xv : alpha[m]*xv;
        Db[(long)m*sDm + bn] = xv;
      }
    }
  }
}

// ---------------- depthwise 3x3 (pad=dil) + affine + PReLU --------------
__global__ void k_dwconv3(const float* __restrict__ x, const float* __restrict__ w,
                          const float* __restrict__ scale, const float* __restrict__ shift,
                          const float* __restrict__ alpha, float* __restrict__ y,
                          int C, int H, int W, int dil, long total)
{
  long idx = (long)blockIdx.x*blockDim.x + threadIdx.x;
  if (idx >= total) return;
  int HW = H*W;
  int p  = (int)(idx % HW);
  int c  = (int)((idx / HW) % C);
  int ox = p % W, oy = p / W;
  const float* xb = x + (idx - p);   // start of this (b,c) plane
  const float* wc = w + c*9;
  float s = 0.f;
#pragma unroll
  for (int i = 0; i < 3; ++i) {
    int iy = oy + (i-1)*dil;
    if ((unsigned)iy >= (unsigned)H) continue;
#pragma unroll
    for (int j = 0; j < 3; ++j) {
      int ix = ox + (j-1)*dil;
      if ((unsigned)ix >= (unsigned)W) continue;
      s += xb[iy*W+ix] * wc[i*3+j];
    }
  }
  s = s*scale[c] + shift[c];
  y[idx] = (s >= 0.f) ? s : alpha[c]*s;
}

// ------------- per-sample dynamic depthwise 5x5 @ dil 1,2,3 -------------
__global__ void k_dynconv(const float* __restrict__ x, const float* __restrict__ kk,
                          float* __restrict__ y, int C, int H, int W, long total)
{
  long idx = (long)blockIdx.x*blockDim.x + threadIdx.x;
  if (idx >= total) return;
  int HW = H*W;
  int p  = (int)(idx % HW);
  long t = idx / HW;
  int c  = (int)(t % C);
  int b  = (int)(t / C);
  if (b == 0) { y[idx] = x[idx]; return; }   // sample 0 passthrough
  int ox = p % W, oy = p / W;
  const float* xb = x + (idx - p);
  const float* kp = kk + ((long)b*C + c)*25;
  float s = 0.f;
  for (int d = 1; d <= 3; ++d)
    for (int i = 0; i < 5; ++i) {
      int iy = oy + (i-2)*d;
      if ((unsigned)iy >= (unsigned)H) continue;
      for (int j = 0; j < 5; ++j) {
        int ix = ox + (j-2)*d;
        if ((unsigned)ix >= (unsigned)W) continue;
        s += xb[iy*W+ix] * kp[i*5+j];
      }
    }
  y[idx] = s;
}

// --------------- bilinear resize, align_corners=True --------------------
__global__ void k_resize(const float* __restrict__ x, float* __restrict__ y,
                         int C, int H, int W, int OH, int OW,
                         long sXb, long sXc, long sYb, long sYc, long total)
{
  long idx = (long)blockIdx.x*blockDim.x + threadIdx.x;
  if (idx >= total) return;
  int ox = (int)(idx % OW);
  long t = idx / OW;
  int oy = (int)(t % OH); t /= OH;
  int c  = (int)(t % C);
  int b  = (int)(t / C);
  int y0, y1, x0, x1; float fy, fx;
  if (OH == 1 || H == 1) { y0 = y1 = 0; fy = 0.f; }
  else {
    float ty = oy * ((float)(H-1)/(float)(OH-1));
    y0 = (int)floorf(ty); if (y0 > H-2) y0 = H-2; y1 = y0+1; fy = ty - y0;
  }
  if (OW == 1 || W == 1) { x0 = x1 = 0; fx = 0.f; }
  else {
    float tx = ox * ((float)(W-1)/(float)(OW-1));
    x0 = (int)floorf(tx); if (x0 > W-2) x0 = W-2; x1 = x0+1; fx = tx - x0;
  }
  const float* xp = x + (long)b*sXb + (long)c*sXc;
  float top = xp[y0*W+x0]*(1.f-fx) + xp[y0*W+x1]*fx;
  float bot = xp[y1*W+x0]*(1.f-fx) + xp[y1*W+x1]*fx;
  y[(long)b*sYb + (long)c*sYc + (long)oy*OW + ox] = top*(1.f-fy) + bot*fy;
}

// --------------- adaptive avg pool -> 5x5 -------------------------------
__global__ void k_apool5(const float* __restrict__ x, float* __restrict__ y,
                         int H, int W, long total)
{
  long idx = (long)blockIdx.x*blockDim.x + threadIdx.x;
  if (idx >= total) return;
  int j  = (int)(idx % 5);
  long t = idx / 5;
  int i  = (int)(t % 5);
  long bc = t / 5;
  int ys = (i*H)/5,   ye = ((i+1)*H + 4)/5;
  int xs = (j*W)/5,   xe = ((j+1)*W + 4)/5;
  const float* xp = x + bc*(long)H*W;
  float s = 0.f;
  for (int yy = ys; yy < ye; ++yy)
    for (int xx = xs; xx < xe; ++xx) s += xp[yy*W+xx];
  y[idx] = s / (float)((ye-ys)*(xe-xs));
}

// -------- edge = x - avgpool3(x)/9 ; also emit prelu(edge) --------------
__global__ void k_edge(const float* __restrict__ x, const float* __restrict__ alpha,
                       float* __restrict__ edge, float* __restrict__ edge_prelu,
                       int C, int H, int W, long total)
{
  long idx = (long)blockIdx.x*blockDim.x + threadIdx.x;
  if (idx >= total) return;
  int HW = H*W;
  int p  = (int)(idx % HW);
  int c  = (int)((idx / HW) % C);
  int ox = p % W, oy = p / W;
  const float* xb = x + (idx - p);
  float s = 0.f;
#pragma unroll
  for (int i = -1; i <= 1; ++i) {
    int iy = oy + i;
    if ((unsigned)iy >= (unsigned)H) continue;
#pragma unroll
    for (int j = -1; j <= 1; ++j) {
      int ix = ox + j;
      if ((unsigned)ix >= (unsigned)W) continue;
      s += xb[iy*W+ix];
    }
  }
  float e = x[idx] - s*(1.f/9.f);
  edge[idx] = e;
  edge_prelu[idx] = (e >= 0.f) ? e : alpha[c]*e;
}

// --------------- out = sigmoid(wv)*x + x --------------------------------
__global__ void k_gate(const float* __restrict__ wv, const float* __restrict__ x,
                       float* __restrict__ y, long total)
{
  long idx = (long)blockIdx.x*blockDim.x + threadIdx.x;
  if (idx >= total) return;
  float s = 1.f/(1.f + expf(-wv[idx]));
  y[idx] = s*x[idx] + x[idx];
}

__global__ void k_add(const float* __restrict__ a, const float* __restrict__ b,
                      float* __restrict__ y, long total)
{
  long idx = (long)blockIdx.x*blockDim.x + threadIdx.x;
  if (idx < total) y[idx] = a[idx] + b[idx];
}

__global__ void k_sigmoid(const float* __restrict__ x, float* __restrict__ y, long total)
{
  long idx = (long)blockIdx.x*blockDim.x + threadIdx.x;
  if (idx < total) y[idx] = 1.f/(1.f + expf(-x[idx]));
}

// ----- salhead: y[b,n] = b0 + sum_c w[c]*x[b,c,n] (strided x) -----------
__global__ void k_salhead(const float* __restrict__ x, const float* __restrict__ w,
                          const float* __restrict__ b0, float* __restrict__ y,
                          int C, int HW, long sXb, long sXc, long total)
{
  long idx = (long)blockIdx.x*blockDim.x + threadIdx.x;
  if (idx >= total) return;
  int n = (int)(idx % HW);
  int b = (int)(idx / HW);
  const float* xp = x + (long)b*sXb + n;
  float s = b0[0];
  for (int c = 0; c < C; ++c) s += w[c]*xp[(long)c*sXc];
  y[idx] = s;
}

// ---- softmax: out[b,i,j] = softmax_j( A[b*C*C + i*rs + j*cs] ), one wave per (b,i)
__global__ __launch_bounds__(32)
void k_softmax(const float* __restrict__ A, float* __restrict__ out,
               int C, long rs, long cs)
{
  int b = blockIdx.z, i = blockIdx.x, lane = threadIdx.x;
  const float* Ab = A + (long)b*C*C;
  float m = -3.4e38f;
  for (int j = lane; j < C; j += 32) m = fmaxf(m, Ab[(long)i*rs + (long)j*cs]);
  for (int o = 16; o; o >>= 1) m = fmaxf(m, __shfl_xor(m, o, 32));
  float s = 0.f;
  for (int j = lane; j < C; j += 32) s += expf(Ab[(long)i*rs + (long)j*cs] - m);
  for (int o = 16; o; o >>= 1) s += __shfl_xor(s, o, 32);
  float inv = 1.f/s;
  float* ob = out + (long)b*C*C + (long)i*C;
  for (int j = lane; j < C; j += 32)
    ob[j] = expf(Ab[(long)i*rs + (long)j*cs] - m) * inv;
}

// =====================================================================
// Host orchestration
// =====================================================================
struct CBR { const float *alpha, *bias, *scale, *w; };

static inline dim3 eg(long n) { return dim3((unsigned)((n + 255)/256)); }

static void gemm(hipStream_t st, const float* A, const float* B, float* D,
                 int Bb, int M, int N, int K,
                 long sAb, long sAm, long sAk, long sBb, long sBk, long sBn,
                 long sDb, long sDm,
                 const float* bias0, const float* scale, const float* shift,
                 const float* alpha)
{
  dim3 g(DIVUP(DIVUP(N,16),4), DIVUP(M,16), Bb);
  k_gemm_wmma<<<g, dim3(32,4), 0, st>>>(A, B, D, M, N, K, sAb,sAm,sAk,
                                        sBb,sBk,sBn, sDb,sDm,
                                        bias0, scale, shift, alpha);
}

static void dwconv(hipStream_t st, const float* x, CBR p, float* y,
                   int B, int C, int H, int W)
{
  long n = (long)B*C*H*W;
  k_dwconv3<<<eg(n),256,0,st>>>(x, p.w, p.scale, p.bias, p.alpha, y, C, H, W, 1, n);
}

static void pwconv(hipStream_t st, CBR p, const float* x, float* d,
                   int B, int Cout, int Cin, int HW, long sBb, long sDb, long sDm)
{
  gemm(st, p.w, x, d, B, Cout, HW, Cin, 0, Cin, 1, sBb, HW, 1, sDb, sDm,
       nullptr, p.scale, p.bias, p.alpha);
}

static void dsconv(hipStream_t st, const float* x, float* tmp, float* out,
                   CBR dw, CBR pw, int B, int Cin, int Cout, int H, int W,
                   long sDb, long sDm)
{
  dwconv(st, x, dw, tmp, B, Cin, H, W);
  pwconv(st, pw, tmp, out, B, Cout, Cin, H*W, (long)Cin*H*W, sDb, sDm);
}

static void resize(hipStream_t st, const float* x, float* y, int B, int C,
                   int H, int W, int OH, int OW, long sXb, long sXc)
{
  long n = (long)B*C*OH*OW;
  k_resize<<<eg(n),256,0,st>>>(x, y, C, H, W, OH, OW, sXb, sXc,
                               (long)C*OH*OW, (long)OH*OW, n);
}

extern "C" void kernel_launch(void* const* d_in, const int* in_sizes, int n_in,
                              void* d_out, int out_size, void* d_ws, size_t ws_size,
                              hipStream_t stream)
{
  (void)in_sizes; (void)n_in; (void)out_size; (void)ws_size;
  hipStream_t st = stream;
#define P(i) ((const float*)d_in[(i)])
  // d_in layout (jax pytree flatten: dict keys sorted, lists in order):
  // 0..4: conv1..conv5. 5..12: conv5_conv3 ds. 13..20: conv5_conv4 ds.
  // dec: d12 21..44, d34 45..68, d5 69..92, s12 93/94, s34 95/96, s5 97/98.
  // dsmm: ccorr(conv1 99..106, linear_e 107), fuse3 108..111, fuse4 112..115,
  //       smooth3 116..123, smooth4 124..131.
  // esam: ccorr(conv1 132..139, linear_e 140), eeu1 141..145, eeu2 146..150,
  //       smooth1 151..158, smooth2 159..166.
  auto cb = [&](int i)->CBR { return CBR{P(i), P(i+1), P(i+2), P(i+3)}; };

  const int B = 8;
  const int HW5 = 64, HW4 = 256, HW3 = 1024, HW1 = 16384, HW0 = 65536;

  float* ws = (float*)d_ws;
  float* out = (float*)d_out;
  size_t o = 0;
  auto A_ = [&](size_t n){ size_t r = o; o += n; return r; };

  // --- persistent / big buffers (cat12+a12+b12 exactly overlays x12d) ---
  size_t o_cat54 = A_((size_t)B*384*HW3);       // x5d | x3_out | x4_out
  size_t o_cat12 = A_((size_t)B*96*HW1);        // x34d | x1_out | x2_out
  size_t o_a12   = A_((size_t)B*48*HW1);
  size_t o_b12   = A_((size_t)B*48*HW1);
  size_t o_x12d  = o_cat12;                     // alias: cat12+a12+b12 == B*48*HW0
  size_t o_slabA = A_((size_t)B*48*HW0);        // 25.2M: r34 / d12 temps / r12
  size_t o_slabB = A_((size_t)B*48*HW0);        // 25.2M: t34 / t12
  size_t o_s5    = A_((size_t)B*HW3);
  size_t o_s34   = A_((size_t)B*HW1);
  // --- small phase ---
  size_t o_t320 = A_((size_t)B*320*HW5);
  size_t o_pw96 = A_((size_t)B*96*HW5);
  size_t o_kc4  = A_((size_t)B*96*25);
  size_t o_pw32 = A_((size_t)B*32*HW5);
  size_t o_kc3  = A_((size_t)B*32*25);
  size_t o_dyn4 = A_((size_t)B*96*HW4);
  size_t o_x4al = A_((size_t)B*96*HW4);
  size_t o_x4up = A_((size_t)B*96*HW3);
  size_t o_t96  = A_((size_t)B*96*HW3);
  size_t o_x4s  = A_((size_t)B*96*HW3);
  size_t o_dyn3 = A_((size_t)B*32*HW3);
  size_t o_f3   = A_((size_t)B*32*HW3);
  size_t o_t32  = A_((size_t)B*32*HW3);
  size_t o_x3s  = A_((size_t)B*96*HW3);
  size_t o_g96  = A_((size_t)B*96*96);
  size_t o_A96  = A_((size_t)B*96*96);
  size_t o_A196 = A_((size_t)B*96*96);
  size_t o_Bm96 = A_((size_t)B*96*96);
  size_t o_qa96 = A_((size_t)B*96*HW3);
  size_t o_ea96 = A_((size_t)B*96*HW3);
  size_t o_ei96 = A_((size_t)B*96*HW3);
  size_t o_qi96 = A_((size_t)B*96*HW3);
  // --- dec small ---
  size_t o_d5a  = A_((size_t)B*320*HW5);
  size_t o_d5b  = A_((size_t)B*320*HW5);
  size_t o_r5   = A_((size_t)B*320*HW3);
  size_t o_t5   = A_((size_t)B*320*HW3);
  size_t o_a34  = A_((size_t)B*192*HW3);
  size_t o_b34  = A_((size_t)B*192*HW3);
  size_t o_t384 = A_((size_t)B*384*HW3);
  size_t o_t192 = A_((size_t)B*192*HW3);
  // --- esam ---
  size_t o_SA  = A_((size_t)B*24*HW1);
  size_t o_SB  = A_((size_t)B*24*HW1);
  size_t o_SC  = A_((size_t)B*24*HW1);
  size_t o_SD  = A_((size_t)B*24*HW1);
  size_t o_SE  = A_((size_t)B*24*HW1);
  size_t o_t24 = A_((size_t)B*24*HW1);
  size_t o_t16 = A_((size_t)B*16*HW1);
  size_t o_g24  = A_((size_t)B*24*24);
  size_t o_A24  = A_((size_t)B*24*24);
  size_t o_A124 = A_((size_t)B*24*24);
  size_t o_Bm24 = A_((size_t)B*24*24);

  // output slots
  const size_t O_S12 = 0, O_S34U = 524288, O_S5U = 1048576,
               O_G12 = 1572864, O_G34 = 2097152, O_G5 = 2621440,
               O_E1 = 3145728, O_E2 = 6291456;

  // =================== conv5 -> kernels kc4 / kc3 ======================
  dsconv(st, P(4), ws+o_t320, ws+o_pw96, cb(13), cb(17), B, 320, 96, 8, 8,
         (long)96*HW5, HW5);
  { long n = (long)B*96*25; k_apool5<<<eg(n),256,0,st>>>(ws+o_pw96, ws+o_kc4, 8, 8, n); }
  dsconv(st, P(4), ws+o_t320, ws+o_pw32, cb(5), cb(9), B, 320, 32, 8, 8,
         (long)32*HW5, HW5);
  { long n = (long)B*32*25; k_apool5<<<eg(n),256,0,st>>>(ws+o_pw32, ws+o_kc3, 8, 8, n); }

  // =========================== DSMM ====================================
  { long n = (long)B*96*HW4; k_dynconv<<<eg(n),256,0,st>>>(P(3), ws+o_kc4, ws+o_dyn4, 96, 16, 16, n); }
  { CBR f4 = cb(112);
    pwconv(st, f4, ws+o_dyn4, ws+o_x4al, B, 96, 96, HW4, (long)96*HW4, (long)96*HW4, HW4); }
  resize(st, ws+o_x4al, ws+o_x4up, B, 96, 16, 16, 32, 32, (long)96*HW4, HW4);
  dsconv(st, ws+o_x4up, ws+o_t96, ws+o_x4s, cb(124), cb(128), B, 96, 96, 32, 32,
         (long)96*HW3, HW3);
  { long n = (long)B*32*HW3; k_dynconv<<<eg(n),256,0,st>>>(P(2), ws+o_kc3, ws+o_dyn3, 32, 32, 32, n); }
  { CBR f3 = cb(108);
    pwconv(st, f3, ws+o_dyn3, ws+o_f3, B, 32, 32, HW3, (long)32*HW3, (long)32*HW3, HW3); }
  dsconv(st, ws+o_f3, ws+o_t32, ws+o_x3s, cb(116), cb(120), B, 32, 96, 32, 32,
         (long)96*HW3, HW3);

  // ccorrm(exemplar=x3s, query=x4s), C=96, N=1024 (resize is identity here)
  gemm(st, ws+o_x4s, ws+o_x3s, ws+o_g96, B, 96, 96, HW3,
       (long)96*HW3, HW3, 1,   (long)96*HW3, 1, HW3,   96*96, 96,
       nullptr,nullptr,nullptr,nullptr);                               // Gram q.ex^T
  gemm(st, ws+o_g96, P(107), ws+o_A96, B, 96, 96, 96,
       96*96, 96, 1,   0, 1, 96,   96*96, 96,
       nullptr,nullptr,nullptr,nullptr);                               // A = M.linear_e^T
  k_softmax<<<dim3(96,1,B),32,0,st>>>(ws+o_A96, ws+o_A196, 96, 96, 1); // row softmax
  k_softmax<<<dim3(96,1,B),32,0,st>>>(ws+o_A96, ws+o_Bm96, 96, 1, 96); // col softmax -> Bm[d,c]
  gemm(st, ws+o_A196, ws+o_x3s, ws+o_qa96, B, 96, HW3, 96,
       96*96, 96, 1,   (long)96*HW3, HW3, 1,   (long)96*HW3, HW3,
       nullptr,nullptr,nullptr,nullptr);                               // query_att
  gemm(st, ws+o_Bm96, ws+o_x4s, ws+o_ea96, B, 96, HW3, 96,
       96*96, 96, 1,   (long)96*HW3, HW3, 1,   (long)96*HW3, HW3,
       nullptr,nullptr,nullptr,nullptr);                               // ex_att
  { long n = (long)B*96*HW3;
    k_add<<<eg(n),256,0,st>>>(ws+o_ea96, ws+o_x3s, ws+o_ei96, n);
    k_add<<<eg(n),256,0,st>>>(ws+o_qa96, ws+o_x4s, ws+o_qi96, n); }
  // conv1 shared for both branches; write straight into cat54 channel slices
  dsconv(st, ws+o_ei96, ws+o_t96, ws + o_cat54 + (size_t)192*HW3,
         cb(99), cb(103), B, 96, 96, 32, 32, (long)384*HW3, HW3);      // x3_out
  dsconv(st, ws+o_qi96, ws+o_t96, ws + o_cat54 + (size_t)288*HW3,
         cb(99), cb(103), B, 96, 96, 32, 32, (long)384*HW3, HW3);      // x4_out

  // =========================== ESAM ====================================
  dsconv(st, P(0), ws+o_t16, ws+o_SA, cb(151), cb(155), B, 16, 24, 128, 128,
         (long)24*HW1, HW1);                                           // s1
  { long n = (long)B*24*HW1;
    k_edge<<<eg(n),256,0,st>>>(ws+o_SA, P(141), ws+o_SB, out+O_E1, 24, 128, 128, n); }
  gemm(st, P(145), ws+o_SB, ws+o_SC, B, 24, HW1, 24,
       0, 24, 1,   (long)24*HW1, HW1, 1,   (long)24*HW1, HW1,
       P(142), P(144), P(143), nullptr);                               // eeu1 gate conv
  { long n = (long)B*24*HW1; k_gate<<<eg(n),256,0,st>>>(ws+o_SC, ws+o_SA, ws+o_SD, n); } // x1_2
  resize(st, P(1), ws+o_SA, B, 24, 64, 64, 128, 128, (long)24*4096, 4096);               // x2u
  dsconv(st, ws+o_SA, ws+o_t24, ws+o_SB, cb(159), cb(163), B, 24, 24, 128, 128,
         (long)24*HW1, HW1);                                           // s2
  { long n = (long)B*24*HW1;
    k_edge<<<eg(n),256,0,st>>>(ws+o_SB, P(146), ws+o_SC, out+O_E2, 24, 128, 128, n); }
  gemm(st, P(150), ws+o_SC, ws+o_SA, B, 24, HW1, 24,
       0, 24, 1,   (long)24*HW1, HW1, 1,   (long)24*HW1, HW1,
       P(147), P(149), P(148), nullptr);                               // eeu2 gate conv
  { long n = (long)B*24*HW1; k_gate<<<eg(n),256,0,st>>>(ws+o_SA, ws+o_SB, ws+o_SE, n); } // x2_2

  // ccorrm(exemplar=x1_2(SD), query=x2_2(SE)), C=24, N=16384
  gemm(st, ws+o_SE, ws+o_SD, ws+o_g24, B, 24, 24, HW1,
       (long)24*HW1, HW1, 1,   (long)24*HW1, 1, HW1,   24*24, 24,
       nullptr,nullptr,nullptr,nullptr);                               // Gram q.ex^T
  gemm(st, ws+o_g24, P(140), ws+o_A24, B, 24, 24, 24,
       24*24, 24, 1,   0, 1, 24,   24*24, 24, nullptr,nullptr,nullptr,nullptr);
  k_softmax<<<dim3(24,1,B),32,0,st>>>(ws+o_A24, ws+o_A124, 24, 24, 1);
  k_softmax<<<dim3(24,1,B),32,0,st>>>(ws+o_A24, ws+o_Bm24, 24, 1, 24);
  gemm(st, ws+o_A124, ws+o_SD, ws+o_SA, B, 24, HW1, 24,
       24*24, 24, 1,   (long)24*HW1, HW1, 1,   (long)24*HW1, HW1,
       nullptr,nullptr,nullptr,nullptr);                               // query_att
  gemm(st, ws+o_Bm24, ws+o_SE, ws+o_SB, B, 24, HW1, 24,
       24*24, 24, 1,   (long)24*HW1, HW1, 1,   (long)24*HW1, HW1,
       nullptr,nullptr,nullptr,nullptr);                               // ex_att
  { long n = (long)B*24*HW1;
    k_add<<<eg(n),256,0,st>>>(ws+o_SB, ws+o_SD, ws+o_SC, n);           // ex_in
    k_add<<<eg(n),256,0,st>>>(ws+o_SA, ws+o_SE, ws+o_SD, n); }         // q_in
  dsconv(st, ws+o_SC, ws+o_t24, ws + o_cat12 + (size_t)48*HW1,
         cb(132), cb(136), B, 24, 24, 128, 128, (long)96*HW1, HW1);    // x1_out
  dsconv(st, ws+o_SD, ws+o_t24, ws + o_cat12 + (size_t)72*HW1,
         cb(132), cb(136), B, 24, 24, 128, 128, (long)96*HW1, HW1);    // x2_out

  // =========================== decoder =================================
  // d5
  dsconv(st, P(4), ws+o_t320, ws+o_d5a, cb(69), cb(73), B, 320, 320, 8, 8,
         (long)320*HW5, HW5);
  dsconv(st, ws+o_d5a, ws+o_t320, ws+o_d5b, cb(77), cb(81), B, 320, 320, 8, 8,
         (long)320*HW5, HW5);
  resize(st, ws+o_d5b, ws+o_r5, B, 320, 8, 8, 32, 32, (long)320*HW5, HW5);
  dsconv(st, ws+o_r5, ws+o_t5, ws+o_cat54, cb(85), cb(89), B, 320, 192, 32, 32,
         (long)384*HW3, HW3);                                          // x5d -> cat54[0:192]
  { long n = (long)B*HW3;
    k_salhead<<<eg(n),256,0,st>>>(ws+o_cat54, P(98), P(97), ws+o_s5,
                                  192, HW3, (long)384*HW3, HW3, n); }  // s5
  // d34
  dsconv(st, ws+o_cat54, ws+o_t384, ws+o_a34, cb(45), cb(49), B, 384, 192, 32, 32,
         (long)192*HW3, HW3);
  dsconv(st, ws+o_a34, ws+o_t192, ws+o_b34, cb(53), cb(57), B, 192, 192, 32, 32,
         (long)192*HW3, HW3);
  resize(st, ws+o_b34, ws+o_slabA, B, 192, 32, 32, 128, 128, (long)192*HW3, HW3);
  dsconv(st, ws+o_slabA, ws+o_slabB, ws+o_cat12, cb(61), cb(65), B, 192, 48, 128, 128,
         (long)96*HW1, HW1);                                           // x34d -> cat12[0:48]
  { long n = (long)B*HW1;
    k_salhead<<<eg(n),256,0,st>>>(ws+o_cat12, P(96), P(95), ws+o_s34,
                                  48, HW1, (long)96*HW1, HW1, n); }    // s34
  // d12
  dsconv(st, ws+o_cat12, ws+o_slabA, ws+o_a12, cb(21), cb(25), B, 96, 48, 128, 128,
         (long)48*HW1, HW1);
  dsconv(st, ws+o_a12, ws+o_slabB, ws+o_b12, cb(29), cb(33), B, 48, 48, 128, 128,
         (long)48*HW1, HW1);
  resize(st, ws+o_b12, ws+o_slabA, B, 48, 128, 128, 256, 256, (long)48*HW1, HW1);
  dsconv(st, ws+o_slabA, ws+o_slabB, ws+o_x12d, cb(37), cb(41), B, 48, 48, 256, 256,
         (long)48*HW0, HW0);                                           // x12d (aliases cat12 region)
  { long n = (long)B*HW0;
    k_salhead<<<eg(n),256,0,st>>>(ws+o_x12d, P(94), P(93), out+O_S12,
                                  48, HW0, (long)48*HW0, HW0, n); }    // s12 -> output

  // =========================== outputs =================================
  resize(st, ws+o_s34, out+O_S34U, B, 1, 128, 128, 256, 256, HW1, HW1); // s34u
  resize(st, ws+o_s5,  out+O_S5U,  B, 1, 32, 32, 256, 256, HW3, HW3);   // s5u
  { long n = (long)B*HW0;
    k_sigmoid<<<eg(n),256,0,st>>>(out+O_S12,  out+O_G12, n);
    k_sigmoid<<<eg(n),256,0,st>>>(out+O_S34U, out+O_G34, n);
    k_sigmoid<<<eg(n),256,0,st>>>(out+O_S5U,  out+O_G5,  n); }
#undef P
}
